// GAT_Simple_55989193671328
// MI455X (gfx1250) — matrix operand order repaired and checked by reading, not compiled, not run on hardware
//
#include <hip/hip_runtime.h>
#include <math.h>

// ---------------- problem constants ----------------
#define N_NODES 50000
#define N_EDGES 800000
#define ET      (N_EDGES + N_NODES)   // edges + self loops = 850000
#define IN_F    128
#define TDIM    64
#define HEADS   4
#define C0      64
#define C1      16
#define HC0     (HEADS * C0)          // 256
#define HC1     (HEADS * C1)          // 64
#define SLOPE   0.2f

typedef __attribute__((ext_vector_type(16))) _Float16 v16h;
typedef __attribute__((ext_vector_type(8)))  float    v8f;

static inline int cdiv(int a, int b) { return (a + b - 1) / b; }

// order-preserving float <-> uint encoding for atomicMax-based segment max
__device__ __forceinline__ unsigned fenc(float f) {
  unsigned u = __float_as_uint(f);
  return (u & 0x80000000u) ? ~u : (u | 0x80000000u);
}
__device__ __forceinline__ float fdec(unsigned u) {
  unsigned b = (u & 0x80000000u) ? (u & 0x7fffffffu) : ~u;
  return __uint_as_float(b);
}

// ---------------- tiny precompute: Ve[td,h] = sum_c lin_edge[td, h*C+c]*att_edge[h,c]
__global__ void k_ve(const float* __restrict__ lin_e, const float* __restrict__ att_e,
                     float* __restrict__ Ve, int C, int HC) {
  int i = blockIdx.x * blockDim.x + threadIdx.x;      // i in [0, TDIM*HEADS)
  if (i >= TDIM * HEADS) return;
  int td = i / HEADS, h = i % HEADS;
  float s = 0.f;
  for (int c = 0; c < C; ++c) s += lin_e[td * HC + h * C + c] * att_e[h * C + c];
  Ve[td * HEADS + h] = s;
}

// ---------------- register-blocked WMMA GEMM: C[MxNCOL] = A[MxK] * B[KxNCOL]
// f32 in, f16 compute, f32 accumulate. Block = 8 waves. Each wave owns one
// 16-row M-stripe and WPT (<=8) N-tiles, so the accumulator set stays at
// <=64 VGPRs (no scratch spills); wide-N GEMMs split each M-tile across WPM
// waves. B is staged per 32-K-step into LDS as f16, column-major, so a B
// fragment is two contiguous ds_load_b128. M multiple of 16; K multiple of 32.
template <int K, int NCOL>
__launch_bounds__(256, 1)
__global__ void k_gemm_wmma(const float* __restrict__ A, const float* __restrict__ B,
                            float* __restrict__ C, int M) {
  constexpr int NT_ALL = NCOL / 16;                   // N-tiles total
  constexpr int WPT    = (NT_ALL > 8) ? 8 : NT_ALL;   // N-tiles per wave (reg-block)
  constexpr int WPM    = NT_ALL / WPT;                // waves sharing one M-tile
  constexpr int MPB    = 8 / WPM;                     // M-tiles per block
  __shared__ _Float16 ldsB[32 * NCOL];                // [n][k] column-major

  int wave = threadIdx.x >> 5;
  int lane = threadIdx.x & 31;
  int mtiles = M >> 4;
  int tm     = blockIdx.x * MPB + wave / WPM;
  int nslice = wave % WPM;
  int tmc = (tm < mtiles) ? tm : (mtiles - 1);        // clamp: tail waves redo a tile
  int r16 = lane & 15;
  int kg  = lane >> 4;
  int row = tmc * 16 + r16;

  v8f zero = {};
  v8f acc[WPT];
#pragma unroll
  for (int t = 0; t < WPT; ++t) acc[t] = zero;

  for (int kk = 0; kk < K; kk += 32) {
    // cooperative stage of B[kk..kk+32)[0..NCOL) -> LDS f16 column-major
    for (int j = threadIdx.x; j < 32 * NCOL; j += 256) {
      int k = j / NCOL, n = j % NCOL;                 // coalesced global read
      ldsB[n * 32 + k] = (_Float16)B[(size_t)(kk + k) * NCOL + n];
    }
    __syncthreads();

    // A fragment (contiguous k order, matching B): halves 0..15 -> K=kk+16*kg+i
    v16h a;
    const float* ap = A + (size_t)row * K + kk + kg * 16;
#pragma unroll
    for (int i = 0; i < 8; ++i) {
      a[i]     = (_Float16)ap[i];
      a[8 + i] = (_Float16)ap[8 + i];
    }

#pragma unroll
    for (int t = 0; t < WPT; ++t) {
      int col = (nslice * WPT + t) * 16 + r16;
      // B fragment: half i -> K = kg*16 + i within step, column = col
      v16h b = *(const v16h*)(ldsB + col * 32 + kg * 16);
      acc[t] = __builtin_amdgcn_wmma_f32_16x16x32_f16(false, a, false, b, (short)0,
                                                      acc[t], false, false);
    }
    __syncthreads();
  }

  if (tm < mtiles) {                                  // wave-uniform predicate
#pragma unroll
    for (int t = 0; t < WPT; ++t) {
      int col = (nslice * WPT + t) * 16 + r16;
#pragma unroll
      for (int r = 0; r < 8; ++r)
        C[(size_t)(tm * 16 + kg * 8 + r) * NCOL + col] = acc[t][r];
    }
  }
}

// ---------------- per-(node,head) attention scalars: s = sum_c h[n,h,c] * a[h,c]
__global__ void k_attn_scal(const float* __restrict__ h, const float* __restrict__ asrc,
                            const float* __restrict__ adst, float* __restrict__ ssrc,
                            float* __restrict__ sdst, int C, int HC) {
  int i = blockIdx.x * blockDim.x + threadIdx.x;      // (node, head)
  if (i >= N_NODES * HEADS) return;
  int n = i / HEADS, hd = i % HEADS;
  const float* hp = h + (size_t)n * HC + hd * C;
  const float* as = asrc + hd * C;
  const float* ad = adst + hd * C;
  float s0 = 0.f, s1 = 0.f;
  for (int c = 0; c < C; ++c) {
    float v = hp[c];
    s0 += v * as[c];
    s1 += v * ad[c];
  }
  ssrc[i] = s0;
  sdst[i] = s1;
}

// ---------------- per-edge time-feature dots for both layers (shared cos work)
__global__ void k_efdot(const float* __restrict__ ts, const float* __restrict__ tw,
                        const float* __restrict__ tb, const float* __restrict__ Ve0,
                        const float* __restrict__ Ve1, float* __restrict__ ef0,
                        float* __restrict__ ef1) {
  int e = blockIdx.x * blockDim.x + threadIdx.x;
  if (e >= ET) return;
  float d0[HEADS] = {0.f, 0.f, 0.f, 0.f};
  float d1[HEADS] = {0.f, 0.f, 0.f, 0.f};
  if (e < N_EDGES) {
    float t = ts[e];
    for (int td = 0; td < TDIM; ++td) {
      float f = fabsf(cosf(t * tw[td] + tb[td]));
#pragma unroll
      for (int h = 0; h < HEADS; ++h) {
        d0[h] += f * Ve0[td * HEADS + h];
        d1[h] += f * Ve1[td * HEADS + h];
      }
    }
  } else {
    // self loop: edge_attr filled with 1.0 -> plain column sums of Ve
    for (int td = 0; td < TDIM; ++td) {
#pragma unroll
      for (int h = 0; h < HEADS; ++h) {
        d0[h] += Ve0[td * HEADS + h];
        d1[h] += Ve1[td * HEADS + h];
      }
    }
  }
#pragma unroll
  for (int h = 0; h < HEADS; ++h) {
    ef0[(size_t)e * HEADS + h] = d0[h];
    ef1[(size_t)e * HEADS + h] = d1[h];
  }
}

// ---------------- alpha = leakyrelu(s_src[src] + s_dst[dst] + efdot) + segment max
__global__ void k_alpha(const int* __restrict__ ei, const float* __restrict__ ssrc,
                        const float* __restrict__ sdst, const float* __restrict__ ef,
                        float* __restrict__ alpha, unsigned* __restrict__ menc) {
  int e = blockIdx.x * blockDim.x + threadIdx.x;
  if (e >= ET) return;
  int s, d;
  if (e < N_EDGES) { s = ei[e]; d = ei[N_EDGES + e]; }
  else             { s = d = e - N_EDGES; }
#pragma unroll
  for (int h = 0; h < HEADS; ++h) {
    float al = ssrc[s * HEADS + h] + sdst[d * HEADS + h] + ef[(size_t)e * HEADS + h];
    al = (al >= 0.f) ? al : SLOPE * al;
    alpha[(size_t)e * HEADS + h] = al;
    atomicMax(&menc[d * HEADS + h], fenc(al));
  }
}

// ---------------- ea = exp(alpha - max[dst]) (in place) + segment sum
__global__ void k_expsum(const int* __restrict__ ei, float* __restrict__ alpha,
                         const unsigned* __restrict__ menc, float* __restrict__ ssum) {
  int e = blockIdx.x * blockDim.x + threadIdx.x;
  if (e >= ET) return;
  int d = (e < N_EDGES) ? ei[N_EDGES + e] : e - N_EDGES;
#pragma unroll
  for (int h = 0; h < HEADS; ++h) {
    float m  = fdec(menc[d * HEADS + h]);
    float ea = expf(alpha[(size_t)e * HEADS + h] - m);
    alpha[(size_t)e * HEADS + h] = ea;
    atomicAdd(&ssum[d * HEADS + h], ea);
  }
}

// ---------------- buffer inits
__global__ void k_fill_u32(unsigned* __restrict__ p, unsigned v, int n) {
  int i = blockIdx.x * blockDim.x + threadIdx.x;
  if (i < n) p[i] = v;
}
__global__ void k_fill_f32(float* __restrict__ p, float v, int n) {
  int i = blockIdx.x * blockDim.x + threadIdx.x;
  if (i < n) p[i] = v;
}
__global__ void k_init_bias(float* __restrict__ out, const float* __restrict__ bias, int HC) {
  int i = blockIdx.x * blockDim.x + threadIdx.x;
  if (i < N_NODES * HC) out[i] = bias[i % HC];
}

// ---------------- weighted scatter-add aggregation: out[dst] += h[src] * a
// one block per edge, one thread per channel
__global__ void k_agg(const int* __restrict__ ei, const float* __restrict__ h,
                      const float* __restrict__ ea, const float* __restrict__ ssum,
                      float* __restrict__ out, int HC, int C) {
  int e = blockIdx.x;
  int c = threadIdx.x;                                // [0, HC)
  int s, d;
  if (e < N_EDGES) { s = ei[e]; d = ei[N_EDGES + e]; }
  else             { s = d = e - N_EDGES; }
  const float* hp = h + (size_t)s * HC + c;
  __builtin_prefetch(hp, 0, 0);                       // global_prefetch_b8: warm L2 gather
  int hd = c / C;
  float w = ea[(size_t)e * HEADS + hd] / (ssum[d * HEADS + hd] + 1e-16f);
  atomicAdd(&out[(size_t)d * HC + c], (*hp) * w);
}

// ---------------- host-side orchestration ----------------
extern "C" void kernel_launch(void* const* d_in, const int* in_sizes, int n_in,
                              void* d_out, int out_size, void* d_ws, size_t ws_size,
                              hipStream_t stream) {
  const float* x     = (const float*)d_in[0];
  const int*   ei    = (const int*)  d_in[1];   // [2, E] row-major: [0..E)=src, [E..2E)=dst
  const float* ts    = (const float*)d_in[2];
  const float* tw    = (const float*)d_in[3];
  const float* tb    = (const float*)d_in[4];
  const float* W0    = (const float*)d_in[5];
  const float* asrc0 = (const float*)d_in[6];
  const float* adst0 = (const float*)d_in[7];
  const float* le0   = (const float*)d_in[8];
  const float* ae0   = (const float*)d_in[9];
  const float* b0    = (const float*)d_in[10];
  const float* W1    = (const float*)d_in[11];
  const float* asrc1 = (const float*)d_in[12];
  const float* adst1 = (const float*)d_in[13];
  const float* le1   = (const float*)d_in[14];
  const float* ae1   = (const float*)d_in[15];
  const float* b1    = (const float*)d_in[16];
  float* out = (float*)d_out;

  // workspace carve-out (~160 MB)
  char* ws = (char*)d_ws;
  size_t off = 0;
  auto carve = [&](size_t bytes) -> void* {
    void* p = ws + off;
    off += (bytes + 255) & ~(size_t)255;
    return p;
  };
  float*    h0    = (float*)   carve((size_t)N_NODES * HC0 * 4);   // x @ W0
  float*    out0  = (float*)   carve((size_t)N_NODES * HC0 * 4);   // layer0 output (+bias0)
  float*    h1    = (float*)   carve((size_t)N_NODES * HC1 * 4);   // out0 @ W1
  float*    ssrc  = (float*)   carve((size_t)N_NODES * HEADS * 4);
  float*    sdst  = (float*)   carve((size_t)N_NODES * HEADS * 4);
  float*    Ve0   = (float*)   carve((size_t)TDIM * HEADS * 4);
  float*    Ve1   = (float*)   carve((size_t)TDIM * HEADS * 4);
  float*    ef0   = (float*)   carve((size_t)ET * HEADS * 4);
  float*    ef1   = (float*)   carve((size_t)ET * HEADS * 4);
  float*    alpha = (float*)   carve((size_t)ET * HEADS * 4);      // reused as ea
  unsigned* menc  = (unsigned*)carve((size_t)N_NODES * HEADS * 4);
  float*    ssum  = (float*)   carve((size_t)N_NODES * HEADS * 4);
  (void)ws_size; (void)in_sizes; (void)n_in; (void)out_size;

  const int B = 256;
  const int mtiles = N_NODES / 16;                    // 3125
  // layer 0: NCOL=256 -> 2 waves per M-tile -> 4 M-tiles per block
  const int gemmBlocks0 = cdiv(mtiles, 4);            // 782
  // layer 1: NCOL=64  -> 1 wave per M-tile  -> 8 M-tiles per block
  const int gemmBlocks1 = cdiv(mtiles, 8);            // 391

  // shared precompute
  k_ve<<<1, B, 0, stream>>>(le0, ae0, Ve0, C0, HC0);
  k_ve<<<1, B, 0, stream>>>(le1, ae1, Ve1, C1, HC1);
  k_efdot<<<cdiv(ET, B), B, 0, stream>>>(ts, tw, tb, Ve0, Ve1, ef0, ef1);

  // ---- layer 0 ----
  k_gemm_wmma<IN_F, HC0><<<gemmBlocks0, B, 0, stream>>>(x, W0, h0, N_NODES);
  k_attn_scal<<<cdiv(N_NODES * HEADS, B), B, 0, stream>>>(h0, asrc0, adst0, ssrc, sdst, C0, HC0);
  k_fill_u32<<<cdiv(N_NODES * HEADS, B), B, 0, stream>>>(menc, 0u, N_NODES * HEADS);
  k_fill_f32<<<cdiv(N_NODES * HEADS, B), B, 0, stream>>>(ssum, 0.f, N_NODES * HEADS);
  k_alpha<<<cdiv(ET, B), B, 0, stream>>>(ei, ssrc, sdst, ef0, alpha, menc);
  k_expsum<<<cdiv(ET, B), B, 0, stream>>>(ei, alpha, menc, ssum);
  k_init_bias<<<cdiv(N_NODES * HC0, B), B, 0, stream>>>(out0, b0, HC0);
  k_agg<<<ET, HC0, 0, stream>>>(ei, h0, alpha, ssum, out0, HC0, C0);

  // ---- layer 1 ----
  k_gemm_wmma<HC0, HC1><<<gemmBlocks1, B, 0, stream>>>(out0, W1, h1, N_NODES);
  k_attn_scal<<<cdiv(N_NODES * HEADS, B), B, 0, stream>>>(h1, asrc1, adst1, ssrc, sdst, C1, HC1);
  k_fill_u32<<<cdiv(N_NODES * HEADS, B), B, 0, stream>>>(menc, 0u, N_NODES * HEADS);
  k_fill_f32<<<cdiv(N_NODES * HEADS, B), B, 0, stream>>>(ssum, 0.f, N_NODES * HEADS);
  k_alpha<<<cdiv(ET, B), B, 0, stream>>>(ei, ssrc, sdst, ef1, alpha, menc);
  k_expsum<<<cdiv(ET, B), B, 0, stream>>>(ei, alpha, menc, ssum);
  k_init_bias<<<cdiv(N_NODES * HC1, B), B, 0, stream>>>(out, b1, HC1);
  k_agg<<<ET, HC1, 0, stream>>>(ei, h1, alpha, ssum, out, HC1, C1);
}
